// KANClassifier_53283364274889
// MI455X (gfx1250) — matrix operand order repaired and checked by reading, not compiled, not run on hardware
//
#include <hip/hip_runtime.h>
#include <hip/hip_bf16.h>
#include <stdint.h>

#define IN_DIM   1024
#define HIDDEN   4096
#define CLASSES  1000
#define NBASIS   16
#define BATCH    4096
#define KDIM     (IN_DIM * NBASIS)   // 16384
#define CPAD     1024                // classes padded to multiple of 256

typedef __attribute__((ext_vector_type(16))) __bf16 v16bf;
typedef __attribute__((ext_vector_type(8)))  float  v8f;
typedef int v4i __attribute__((vector_size(16)));   // matches builtin param type

#if __has_builtin(__builtin_amdgcn_global_load_async_to_lds_b128)
#define HAVE_ASYNC_LDS 1
#else
#define HAVE_ASYNC_LDS 0
#endif

template <int N>
__device__ __forceinline__ void wait_async_le() {
#if __has_builtin(__builtin_amdgcn_s_wait_asynccnt)
  __builtin_amdgcn_s_wait_asynccnt(N);
#else
  asm volatile("s_wait_asynccnt %0" :: "i"(N) : "memory");
#endif
}

__device__ __forceinline__ __bf16 to_bf16(float f) {
  unsigned u = __float_as_uint(f);
  unsigned r = (u + 0x7FFFu + ((u >> 16) & 1u)) >> 16;   // round-to-nearest-even
  unsigned short s = (unsigned short)r;
  return __builtin_bit_cast(__bf16, s);
}

__device__ __forceinline__ void storeOut(float* p, float v)  { *p = v; }
__device__ __forceinline__ void storeOut(__bf16* p, float v) { *p = to_bf16(v); }

// ---------------------------------------------------------------------------
// Stage 1: feats[row, f*16+j] = tanh((x[row,f]-centers[f,j])*scales[f,j]) (bf16)
// ---------------------------------------------------------------------------
__global__ __launch_bounds__(256) void expand_feats(
    const float* __restrict__ x, const float* __restrict__ centers,
    const float* __restrict__ scales, __bf16* __restrict__ feats)
{
  int64_t tid = (int64_t)blockIdx.x * 256 + threadIdx.x;   // row*16384 + f*16 + j
  int j   = (int)(tid & (NBASIS - 1));
  int f   = (int)((tid >> 4) & (IN_DIM - 1));
  int row = (int)(tid >> 14);
  float xv = x[row * IN_DIM + f];
  float z  = (xv - centers[f * NBASIS + j]) * scales[f * NBASIS + j];
  z = fminf(fmaxf(z, -10.f), 10.f);
  float e = __expf(2.f * z);              // tanh(z) = (e^2z - 1)/(e^2z + 1)
  feats[tid] = to_bf16((e - 1.f) / (e + 1.f));
}

// ---------------------------------------------------------------------------
// Transpose + f32->bf16 convert: in [R x C] f32 -> out [Cpad x R] bf16
// (rows c >= C of the output are zero-filled; Cpad = gridDim.x * 32)
// ---------------------------------------------------------------------------
__global__ __launch_bounds__(256) void transpose_convert(
    const float* __restrict__ in, __bf16* __restrict__ out, int R, int C)
{
  __shared__ float tile[32][33];
  int c0 = blockIdx.x * 32, r0 = blockIdx.y * 32;
  int tx = threadIdx.x, ty = threadIdx.y;           // block (32, 8)
  for (int i = 0; i < 4; ++i) {
    int r = r0 + ty + i * 8, c = c0 + tx;
    tile[ty + i * 8][tx] = (c < C) ? in[(size_t)r * C + c] : 0.f;
  }
  __syncthreads();
  for (int i = 0; i < 4; ++i) {
    int c = c0 + ty + i * 8, r = r0 + tx;
    out[(size_t)c * R + r] = to_bf16(tile[tx][ty + i * 8]);
  }
}

// ---------------------------------------------------------------------------
// bf16 WMMA GEMM:  C[M x ldc] = act(A[M x K] @ Bt[Npad x K]^T + bias)
//   workgroup tile 64(M) x 256(N), 8 waves, wave tile 32 x 64, K step 32.
//   3-stage async pipeline (global_load_async_to_lds_b128, ASYNCcnt in-order
//   completion => s_wait_asynccnt<=5 retires exactly the oldest tile).
//   Steady-state loop is manually unrolled by 3 so tile->buffer mapping is a
//   compile-time constant (no %3, no inner branch); K is a template param so
//   trip counts are static. LDS is v_wmma_f32_16x16x32_bf16 fragment order:
//     fragment f (16 rows), lane l, halfwords i=0..15 at [f*512 + l*16 + i]
//   A frag: lane = ((k>>3)&1)*16 + (m&15); i = (k>>4)*8 + (k&7)
//   B frag: lane = ((k>>3)&1)*16 + (n&15); i = (k>>4)*8 + (k&7)
//   => any aligned run of 8 consecutive k for fixed row maps to 16 contiguous
//      bytes of one lane's slice: one async b128 global->LDS transfer.
// ---------------------------------------------------------------------------
template <bool SIGMOID, typename OutT, int K>
__global__ __launch_bounds__(256) void wmma_gemm(
    const __bf16* __restrict__ A,    // [M x K] row-major
    const __bf16* __restrict__ Bt,   // [Npad x K] row-major (B transposed)
    const float*  __restrict__ bias, // [Nreal]
    OutT* __restrict__ C,            // [M x ldc]
    int Nreal, int ldc)
{
  constexpr int NT = K / 32;                         // k-tiles
  static_assert(NT >= 3 && (NT - 2) % 3 == 0, "steady-state must unroll by 3");

  __shared__ alignas(32) __bf16 aT[3][4 * 512];      // 3 x 4 KB
  __shared__ alignas(32) __bf16 bT[3][16 * 512];     // 3 x 16 KB

  const int tid  = threadIdx.x;
  const int lane = tid & 31, wave = tid >> 5;
  const int wm = wave & 1;        // 2 waves along M (32 rows each)
  const int wn = wave >> 1;       // 4 waves along N (64 cols each)
  const int mBase = blockIdx.y * 64;
  const int nBase = blockIdx.x * 256;

  // --- A staging: 64 rows x 4 k-quads = 256 slots (1 per thread) ---
  const int aRow = tid >> 2, akq = tid & 3;
  const int aDstOff = (aRow >> 4) * 512 +
                      ((((akq & 1) << 4) | (aRow & 15)) * 16) + (akq >> 1) * 8;
  const __bf16* aSrc = A + (size_t)(mBase + aRow) * K + akq * 8;

  // --- B staging: 256 rows x 4 k-quads = 1024 slots (4 per thread) ---
  int bDstOff[4]; const __bf16* bSrc[4];
#pragma unroll
  for (int s = 0; s < 4; ++s) {
    int slot = tid + s * 256;
    int r = slot >> 2, q = slot & 3;
    bDstOff[s] = (r >> 4) * 512 +
                 ((((q & 1) << 4) | (r & 15)) * 16) + (q >> 1) * 8;
    bSrc[s] = Bt + (size_t)(nBase + r) * K + q * 8;
  }

  v8f acc[2][4] = {};

  // Issue (no wait): 5 async b128 global->LDS transfers per thread-slice.
  auto stage = [&](int kb, int buf) {
#if HAVE_ASYNC_LDS
    __builtin_amdgcn_global_load_async_to_lds_b128(
        (v4i*)(aSrc + kb), (v4i*)&aT[buf][aDstOff], 0, 0);
#pragma unroll
    for (int s = 0; s < 4; ++s)
      __builtin_amdgcn_global_load_async_to_lds_b128(
          (v4i*)(bSrc[s] + kb), (v4i*)&bT[buf][bDstOff[s]], 0, 0);
#else
    uint4 ra = *(const uint4*)(aSrc + kb);
    uint4 rb0 = *(const uint4*)(bSrc[0] + kb);
    uint4 rb1 = *(const uint4*)(bSrc[1] + kb);
    uint4 rb2 = *(const uint4*)(bSrc[2] + kb);
    uint4 rb3 = *(const uint4*)(bSrc[3] + kb);
    *(uint4*)&aT[buf][aDstOff]     = ra;
    *(uint4*)&bT[buf][bDstOff[0]]  = rb0;
    *(uint4*)&bT[buf][bDstOff[1]]  = rb1;
    *(uint4*)&bT[buf][bDstOff[2]]  = rb2;
    *(uint4*)&bT[buf][bDstOff[3]]  = rb3;
#endif
  };

  auto compute = [&](int buf) {
    const __bf16* aB = &aT[buf][0];
    const __bf16* bB = &bT[buf][0];
    v16bf a0 = *(const v16bf*)(aB + (wm * 2 + 0) * 512 + lane * 16);
    v16bf a1 = *(const v16bf*)(aB + (wm * 2 + 1) * 512 + lane * 16);
    v16bf b0 = *(const v16bf*)(bB + (wn * 4 + 0) * 512 + lane * 16);
    v16bf b1 = *(const v16bf*)(bB + (wn * 4 + 1) * 512 + lane * 16);
    v16bf b2 = *(const v16bf*)(bB + (wn * 4 + 2) * 512 + lane * 16);
    v16bf b3 = *(const v16bf*)(bB + (wn * 4 + 3) * 512 + lane * 16);
    acc[0][0] = __builtin_amdgcn_wmma_f32_16x16x32_bf16(false, a0, false, b0, (short)0, acc[0][0], false, false);
    acc[1][0] = __builtin_amdgcn_wmma_f32_16x16x32_bf16(false, a1, false, b0, (short)0, acc[1][0], false, false);
    acc[0][1] = __builtin_amdgcn_wmma_f32_16x16x32_bf16(false, a0, false, b1, (short)0, acc[0][1], false, false);
    acc[1][1] = __builtin_amdgcn_wmma_f32_16x16x32_bf16(false, a1, false, b1, (short)0, acc[1][1], false, false);
    acc[0][2] = __builtin_amdgcn_wmma_f32_16x16x32_bf16(false, a0, false, b2, (short)0, acc[0][2], false, false);
    acc[1][2] = __builtin_amdgcn_wmma_f32_16x16x32_bf16(false, a1, false, b2, (short)0, acc[1][2], false, false);
    acc[0][3] = __builtin_amdgcn_wmma_f32_16x16x32_bf16(false, a0, false, b3, (short)0, acc[0][3], false, false);
    acc[1][3] = __builtin_amdgcn_wmma_f32_16x16x32_bf16(false, a1, false, b3, (short)0, acc[1][3], false, false);
  };

  // Prologue: tiles 0 and 1 in flight; retire tile 0 (in-order completion).
  stage(0, 0);
  stage(32, 1);
#if HAVE_ASYNC_LDS
  wait_async_le<5>();
#endif
  __syncthreads();

  // Steady state: tile i lives in buf i%3; slot pattern (compute, stage-target)
  // = (0,2),(1,0),(2,1) repeating. (NT-2) iterations, unrolled by 3.
  int kb = 0;
  for (int g = 0; g < (NT - 2) / 3; ++g) {
    stage(kb + 64, 2); compute(0);
#if HAVE_ASYNC_LDS
    wait_async_le<5>();
#endif
    __syncthreads(); kb += 32;

    stage(kb + 64, 0); compute(1);
#if HAVE_ASYNC_LDS
    wait_async_le<5>();
#endif
    __syncthreads(); kb += 32;

    stage(kb + 64, 1); compute(2);
#if HAVE_ASYNC_LDS
    wait_async_le<5>();
#endif
    __syncthreads(); kb += 32;
  }

  // Drain: tiles NT-2 (buf 0, already resident) and NT-1 (buf 1, in flight).
  compute(0);
#if HAVE_ASYNC_LDS
  wait_async_le<0>();
#endif
  __syncthreads();
  compute(1);

  // --- Epilogue: D VGPR r -> m = frag*16 + (lane>>4)*8 + r ; n = frag*16 + (lane&15)
  const int mOff = (lane >> 4) * 8;
  const int nOff = lane & 15;
#pragma unroll
  for (int fi = 0; fi < 2; ++fi) {
#pragma unroll
    for (int fj = 0; fj < 4; ++fj) {
      int n = nBase + (wn * 4 + fj) * 16 + nOff;
      if (n >= Nreal) continue;
      float bv = bias[n];
      int m0 = mBase + (wm * 2 + fi) * 16 + mOff;
#pragma unroll
      for (int r = 0; r < 8; ++r) {
        float v = acc[fi][fj][r] + bv;
        if (SIGMOID) v = 1.f / (1.f + __expf(-v));
        storeOut(&C[(size_t)(m0 + r) * ldc + n], v);
      }
    }
  }
}

// ---------------------------------------------------------------------------
extern "C" void kernel_launch(void* const* d_in, const int* in_sizes, int n_in,
                              void* d_out, int out_size, void* d_ws, size_t ws_size,
                              hipStream_t stream)
{
  const float* x       = (const float*)d_in[0];
  const float* centers = (const float*)d_in[1];
  const float* scales  = (const float*)d_in[2];
  const float* Wb      = (const float*)d_in[3];
  const float* bb      = (const float*)d_in[4];
  const float* Wh      = (const float*)d_in[5];
  const float* bh      = (const float*)d_in[6];
  float* out = (float*)d_out;

  char* ws = (char*)d_ws;
  __bf16* feats = (__bf16*)ws;                                   // [BATCH x KDIM]
  size_t off = (size_t)BATCH * KDIM * sizeof(__bf16);
  __bf16* WbT = (__bf16*)(ws + off);                             // [HIDDEN x KDIM]
  off += (size_t)HIDDEN * KDIM * sizeof(__bf16);
  __bf16* phi = (__bf16*)(ws + off);                             // [BATCH x HIDDEN]
  off += (size_t)BATCH * HIDDEN * sizeof(__bf16);
  __bf16* WhT = (__bf16*)(ws + off);                             // [CPAD x HIDDEN]

  // 1) tanh basis expansion -> bf16 feats
  expand_feats<<<((size_t)BATCH * KDIM) / 256, 256, 0, stream>>>(
      x, centers, scales, feats);

  // 2) transpose+convert weights: Wb [KDIM x HIDDEN] -> WbT [HIDDEN x KDIM]
  transpose_convert<<<dim3(HIDDEN / 32, KDIM / 32), dim3(32, 8), 0, stream>>>(
      Wb, WbT, KDIM, HIDDEN);
  //    Wh [HIDDEN x CLASSES] -> WhT [CPAD x HIDDEN] (zero-padded cols)
  transpose_convert<<<dim3(CPAD / 32, HIDDEN / 32), dim3(32, 8), 0, stream>>>(
      Wh, WhT, HIDDEN, CLASSES);

  // 3) GEMM1 + fused sigmoid: phi = sigmoid(feats @ Wb + bb), bf16
  wmma_gemm<true, __bf16, KDIM><<<dim3(HIDDEN / 256, BATCH / 64), 256, 0, stream>>>(
      feats, WbT, bb, phi, HIDDEN, HIDDEN);

  // 4) GEMM2 + bias: logits = phi @ Wh + bh, f32 (N=1000 guarded)
  wmma_gemm<false, float, HIDDEN><<<dim3(CPAD / 256, BATCH / 64), 256, 0, stream>>>(
      phi, WhT, bh, out, CLASSES, CLASSES);
}
